// MultiHeadAttention_25142738551074
// MI455X (gfx1250) — matrix operand order repaired and checked by reading, not compiled, not run on hardware
//
#include <hip/hip_runtime.h>
#include <hip/hip_bf16.h>
#include <math.h>

// Problem constants (match reference)
#define BB   4
#define SS   2048
#define DD   1024
#define HH   16
#define DKK  64
#define DVV  64
// SCALE = 8.0 -> multiply scores by 0.125f

typedef __attribute__((ext_vector_type(16))) _Float16 v16h;
typedef __attribute__((ext_vector_type(8)))  _Float16 v8h;
typedef __attribute__((ext_vector_type(4)))  _Float16 v4h;
typedef __attribute__((ext_vector_type(8)))  float    v8f;
typedef __attribute__((ext_vector_type(4)))  unsigned u32x4;
typedef __attribute__((ext_vector_type(8)))  int      i32x8;
typedef __attribute__((ext_vector_type(4)))  int      i32x4;

static __device__ __forceinline__ v8f wmma_f16(v16h a, v16h b, v8f c) {
  // D = A(16x32 f16) x B(32x16 f16) + C(16x16 f32)
  return __builtin_amdgcn_wmma_f32_16x16x32_f16(false, a, false, b, (short)0, c,
                                                false, false);
}

static __device__ __forceinline__ v16h combine16(v8h lo, v8h hg) {
  v16h t;
#pragma unroll
  for (int e = 0; e < 8; ++e) { t[e] = lo[e]; t[8 + e] = hg[e]; }
  return t;
}

// 32B LDS B-operand fetch as two 16B ds loads (16B-aligned rows).
static __device__ __forceinline__ v16h ld16h(const _Float16* p) {
  return combine16(*(const v8h*)p, *(const v8h*)(p + 8));
}

// ---------------------------------------------------------------------------
// TDM: issue a 2D tensor_load_to_lds (data_size = 2B) with per-row LDS padding.
// rowlen/tile_rows in elements; stride0 = tensor row stride (elements).
// pad_interval_code: pad after 2^(code+1) dwords; pad_amount_code: (code+1) dwords.
// This toolchain exposes the 6-arg builtin: (g0, g1, g2, g3, g4, cpol).
// ---------------------------------------------------------------------------
static __device__ __forceinline__ void tdm_load_2d(const void* gaddr, unsigned lds_byte,
                                                   unsigned rowlen, unsigned tile_rows,
                                                   unsigned tensor_rows, unsigned stride0,
                                                   unsigned pad_i_code, unsigned pad_a_code) {
  unsigned long long ga = (unsigned long long)(size_t)gaddr;
  u32x4 g0;
  g0[0] = 1u;                                   // count=1, user mode, no gather
  g0[1] = lds_byte;                             // lds_addr
  g0[2] = (unsigned)(ga & 0xffffffffu);         // global_addr[31:0]
  g0[3] = (unsigned)(ga >> 32) | 0x80000000u;   // global_addr[56:32] | type=2<<30
  i32x8 g1;
  g1[0] = (int)((1u << 16)                      // data_size = 1 (2 bytes)
                | (1u << 20)                    // pad_enable
                | (pad_i_code << 22) | (pad_a_code << 25));
  g1[1] = (int)((rowlen & 0xffffu) << 16);      // atomic_barrier_addr=0 | tensor_dim0 lo
  g1[2] = (int)((rowlen >> 16) | ((tensor_rows & 0xffffu) << 16));  // dim0 hi | dim1 lo
  g1[3] = (int)((tensor_rows >> 16) | ((rowlen & 0xffffu) << 16));  // dim1 hi | tile_dim0
  g1[4] = (int)(tile_rows & 0xffffu);           // tile_dim1 | tile_dim2=0
  g1[5] = (int)stride0;                         // tensor_dim0_stride[31:0]
  g1[6] = 0;                                    // stride0 hi | stride1 lo (unused, 2D)
  g1[7] = 0;
  i32x4 z4 = {0, 0, 0, 0};
  i32x8 z8 = {0, 0, 0, 0, 0, 0, 0, 0};
  __builtin_amdgcn_tensor_load_to_lds(g0, g1, z4, z4, z8, 0);
}

// ---------------------------------------------------------------------------
// Kernel 0a: W (f32 [K=1024][N=1024]) -> f16 transposed Wt[n][k] (B-operand
// friendly: lane = N, contiguous K halves).
// ---------------------------------------------------------------------------
__global__ __launch_bounds__(256) void wconv_kernel(const float* __restrict__ Wq,
                                                    const float* __restrict__ Wk,
                                                    const float* __restrict__ Wv,
                                                    _Float16* __restrict__ Wt) {
  unsigned idx = blockIdx.x * 256u + threadIdx.x;       // < 3 * 1024 * 1024
  unsigned z   = idx >> 20;
  unsigned rem = idx & ((1u << 20) - 1u);
  unsigned k   = rem >> 10;
  unsigned n   = rem & 1023u;                           // fastest -> coalesced read
  const float* W = (z == 0) ? Wq : ((z == 1) ? Wk : Wv);
  Wt[(size_t)z * 1024 * 1024 + (size_t)n * 1024 + k] = (_Float16)W[k * 1024 + n];
}

// ---------------------------------------------------------------------------
// Kernel 0b: activations q,k,v (f32 [8192][1024]) -> f16 Xh, 4 elems/thread.
// Pure bandwidth pass so the GEMM inner loop is cvt-free.
// ---------------------------------------------------------------------------
__global__ __launch_bounds__(256) void xconv_kernel(const float* __restrict__ xq,
                                                    const float* __restrict__ xk,
                                                    const float* __restrict__ xv,
                                                    _Float16* __restrict__ Xh) {
  size_t idx = ((size_t)blockIdx.x * 256u + threadIdx.x) * 4;  // < 3 * 8M
  unsigned z   = (unsigned)(idx >> 23);
  unsigned off = (unsigned)(idx & ((8u << 20) - 1u));
  const float* x = (z == 0) ? xq : ((z == 1) ? xk : xv);
  float4 f = *(const float4*)(x + off);
  v4h h;
  h[0] = (_Float16)f.x; h[1] = (_Float16)f.y; h[2] = (_Float16)f.z; h[3] = (_Float16)f.w;
  *(v4h*)(Xh + (size_t)z * (8u << 20) + off) = h;
}

// ---------------------------------------------------------------------------
// Kernel 1: QKV projection GEMM Xh[8192,1024](f16) * Wt + b.
// Block = 128 threads = 4 waves; each wave computes a 16x64 strip (4 WMMAs per
// A-operand). grid = (M/16 = 512, N/256 = 4, z = 3).
// Outputs (f16): Qh,Kh as [B][H][S][64]; V transposed as Vt [B][H][64][S].
// ---------------------------------------------------------------------------
__global__ __launch_bounds__(128) void proj_kernel(
    const _Float16* __restrict__ Xh, const _Float16* __restrict__ Wt,
    const float* __restrict__ bq, const float* __restrict__ bk,
    const float* __restrict__ bv, _Float16* __restrict__ Qh,
    _Float16* __restrict__ Kh, _Float16* __restrict__ Vt) {
  const int z = blockIdx.z;
  const _Float16* x    = Xh + (size_t)z * (8u << 20);
  const _Float16* W    = Wt + (size_t)z * 1024 * 1024;
  const float*    bias = (z == 0) ? bq : ((z == 1) ? bk : bv);

  const int lane = threadIdx.x & 31;
  const int wave = threadIdx.x >> 5;
  const int l16  = lane & 15;
  const int hi   = lane >> 4;

  const int mt = blockIdx.x;                       // 16-row tile of flattened B*S
  const int n0 = blockIdx.y * 256 + wave * 64;     // 64-col strip per wave
  const int m  = mt * 16 + l16;                    // A row (M = lane%16)

  const _Float16* arow = x + (size_t)m * DD;
  const _Float16* brow0 = W + (size_t)(n0 + 0  + l16) * DD;
  const _Float16* brow1 = W + (size_t)(n0 + 16 + l16) * DD;
  const _Float16* brow2 = W + (size_t)(n0 + 32 + l16) * DD;
  const _Float16* brow3 = W + (size_t)(n0 + 48 + l16) * DD;

  v8f acc0 = {}, acc1 = {}, acc2 = {}, acc3 = {};
  for (int k0 = 0; k0 < DD; k0 += 32) {
    // A-operand: lanes 0-15 K=0..15, lanes 16-31 K=16..31 (matches B).
    v16h a = *(const v16h*)(arow + k0 + hi * 16);
    // B-operands: lanes 0-15 K=0..15, lanes 16-31 K=16..31 -> contiguous 32B.
    acc0 = wmma_f16(a, *(const v16h*)(brow0 + k0 + hi * 16), acc0);
    acc1 = wmma_f16(a, *(const v16h*)(brow1 + k0 + hi * 16), acc1);
    acc2 = wmma_f16(a, *(const v16h*)(brow2 + k0 + hi * 16), acc2);
    acc3 = wmma_f16(a, *(const v16h*)(brow3 + k0 + hi * 16), acc3);
  }

  const int head = n0 >> 6;                        // whole 64-col strip is one head
#pragma unroll
  for (int j = 0; j < 4; ++j) {
    const v8f& acc = (j == 0) ? acc0 : (j == 1) ? acc1 : (j == 2) ? acc2 : acc3;
    const int n  = n0 + j * 16 + l16;
    const int cn = n & 63;
    const float badd = bias[n];
    if (z < 2) {
      _Float16* dst = (z == 0) ? Qh : Kh;
#pragma unroll
      for (int v = 0; v < 8; ++v) {                // C/D layout: row M = v + 8*hi
        int g  = mt * 16 + v + 8 * hi;             // flattened b*S + s
        int b_ = g >> 11, s_ = g & 2047;
        dst[((size_t)(b_ * HH + head) * SS + s_) * DKK + cn] = (_Float16)(acc[v] + badd);
      }
    } else {
#pragma unroll
      for (int v = 0; v < 8; ++v) {
        int g  = mt * 16 + v + 8 * hi;
        int b_ = g >> 11, s_ = g & 2047;
        Vt[((size_t)(b_ * HH + head) * DVV + cn) * SS + s_] = (_Float16)(acc[v] + badd);
      }
    }
  }
}

// ---------------------------------------------------------------------------
// Kernel 2: flash attention. Block = 256 threads = 8 waves, 16 queries/wave,
// key tiles of 32. K/V tiles staged in LDS by the Tensor Data Mover (wave 0
// issues double-buffered tensor_load_to_lds, TDM pad feature inserts the
// bank-conflict padding), published via s_wait_tensorcnt + barrier.
// grid = (S/128 = 16, H = 16, B = 4).
// ---------------------------------------------------------------------------
__global__ __launch_bounds__(256) void attn_kernel(const _Float16* __restrict__ Qh,
                                                   const _Float16* __restrict__ Kh,
                                                   const _Float16* __restrict__ Vt,
                                                   float* __restrict__ out) {
  // TDM pads K rows 64->72 halves (pad 4 dwords per 32 dwords: codes 4,3) and
  // V rows 32->40 halves (pad 4 dwords per 16 dwords: codes 3,3) so 16-lane
  // B-operand reads hit distinct bank quads (2-way minimum for 32B/lane).
  __shared__ __align__(16) _Float16 ldsK[2][32][72];
  __shared__ __align__(16) _Float16 ldsV[2][64][40];
  __shared__ __align__(16) _Float16 ldsP[8][16][40];

  const int lane = threadIdx.x & 31;
  const int wave = threadIdx.x >> 5;
  const int l16  = lane & 15;
  const int hi   = lane >> 4;

  const int bb = blockIdx.z;
  const int hh = blockIdx.y;
  const int q0 = blockIdx.x * 128 + wave * 16;
  const size_t bh = (size_t)bb * HH + hh;

  const _Float16* Qbase = Qh + bh * SS * DKK;
  const _Float16* Kbase = Kh + bh * SS * DKK;
  const _Float16* Vbase = Vt + bh * DVV * SS;

  // Preload Q as two resident A-operands (K-chunks 0..31 / 32..63).
  v16h qa0, qa1;
  {
    const _Float16* qrow = Qbase + (size_t)(q0 + l16) * DKK;
    qa0 = combine16(*(const v8h*)(qrow + hi * 16), *(const v8h*)(qrow + hi * 16 + 8));
    qa1 = combine16(*(const v8h*)(qrow + 32 + hi * 16), *(const v8h*)(qrow + 32 + hi * 16 + 8));
  }

  v8f o0 = {}, o1 = {}, o2 = {}, o3 = {};
  float mrow[8], lrow[8];
#pragma unroll
  for (int v = 0; v < 8; ++v) { mrow[v] = -INFINITY; lrow[v] = 0.f; }

  // Prologue: stage tile 0 into buffer 0.
  if (wave == 0) {
    tdm_load_2d(Kbase, (unsigned)(size_t)&ldsK[0][0][0], 64, 32, SS, 64, 4, 3);
    tdm_load_2d(Vbase, (unsigned)(size_t)&ldsV[0][0][0], 32, 64, 64, SS, 3, 3);
  }

  for (int kt = 0; kt < SS; kt += 32) {
    const int cur = (kt >> 5) & 1;
    if (wave == 0) {
      if (kt + 32 < SS) {  // prefetch next tile into the other buffer
        tdm_load_2d(Kbase + (size_t)(kt + 32) * DKK,
                    (unsigned)(size_t)&ldsK[cur ^ 1][0][0], 64, 32, SS, 64, 4, 3);
        tdm_load_2d(Vbase + (kt + 32),
                    (unsigned)(size_t)&ldsV[cur ^ 1][0][0], 32, 64, 64, SS, 3, 3);
        __builtin_amdgcn_s_wait_tensorcnt(2);  // current buffer's 2 DMAs done
      } else {
        __builtin_amdgcn_s_wait_tensorcnt(0);
      }
    }
    __syncthreads();  // current tile visible to all 8 waves

    // scores = Q * K^T / 8 for 32 keys (two 16-key C tiles, two DK chunks)
    const _Float16* kr0 = &ldsK[cur][l16][0];
    const _Float16* kr1 = &ldsK[cur][16 + l16][0];
    v8f s0 = {}, s1 = {};
    s0 = wmma_f16(qa0, ld16h(kr0 + hi * 16), s0);
    s0 = wmma_f16(qa1, ld16h(kr0 + 32 + hi * 16), s0);
    s1 = wmma_f16(qa0, ld16h(kr1 + hi * 16), s1);
    s1 = wmma_f16(qa1, ld16h(kr1 + 32 + hi * 16), s1);

    // online softmax (lane = N = key, VGPR v = row M = v + 8*hi;
    // row reductions across the 16-lane group: shfl_xor masks 1,2,4,8)
#pragma unroll
    for (int v = 0; v < 8; ++v) {
      float a = s0[v] * 0.125f;
      float b = s1[v] * 0.125f;
      float t = fmaxf(a, b);
      t = fmaxf(t, __shfl_xor(t, 1, 32));
      t = fmaxf(t, __shfl_xor(t, 2, 32));
      t = fmaxf(t, __shfl_xor(t, 4, 32));
      t = fmaxf(t, __shfl_xor(t, 8, 32));
      float mn = fmaxf(mrow[v], t);
      float p0 = __expf(a - mn);
      float p1 = __expf(b - mn);
      float rs = p0 + p1;
      rs += __shfl_xor(rs, 1, 32);
      rs += __shfl_xor(rs, 2, 32);
      rs += __shfl_xor(rs, 4, 32);
      rs += __shfl_xor(rs, 8, 32);
      float corr = __expf(mrow[v] - mn);
      lrow[v] = lrow[v] * corr + rs;
      mrow[v] = mn;
      o0[v] *= corr; o1[v] *= corr; o2[v] *= corr; o3[v] *= corr;
      int r = v + 8 * hi;
      ldsP[wave][r][l16]      = (_Float16)p0;  // stage P in C layout
      ldsP[wave][r][16 + l16] = (_Float16)p1;
    }
    // wave-internal cross-lane LDS dependency: drain DS before readback
    asm volatile("s_wait_dscnt 0" ::: "memory");

    // P back as A-operand (contiguous 16 keys per lane half)
    const _Float16* pr = &ldsP[wave][l16][0];
    v16h pa = combine16(*(const v8h*)(pr + hi * 16), *(const v8h*)(pr + hi * 16 + 8));

    // O += P(16x32) * V(32x64) from the staged V slice
    o0 = wmma_f16(pa, ld16h(&ldsV[cur][0  + l16][hi * 16]), o0);
    o1 = wmma_f16(pa, ld16h(&ldsV[cur][16 + l16][hi * 16]), o1);
    o2 = wmma_f16(pa, ld16h(&ldsV[cur][32 + l16][hi * 16]), o2);
    o3 = wmma_f16(pa, ld16h(&ldsV[cur][48 + l16][hi * 16]), o3);

    __syncthreads();  // all waves done with buf[cur] before TDM refills it
  }

  // epilogue: normalize by l and store f32 [B,S,H*DV]
#pragma unroll
  for (int v = 0; v < 8; ++v) {
    float inv = 1.0f / lrow[v];
    int sq = q0 + v + 8 * hi;
    size_t rowoff = ((size_t)bb * SS + sq) * DD + (size_t)hh * 64;
    out[rowoff + 0  + l16] = o0[v] * inv;
    out[rowoff + 16 + l16] = o1[v] * inv;
    out[rowoff + 32 + l16] = o2[v] * inv;
    out[rowoff + 48 + l16] = o3[v] * inv;
  }
}

// ---------------------------------------------------------------------------
// Host launcher. Workspace layout (f16):
//   Wt : 3 x 1M                    ( 6 MB) @ +0
//   Xh : 3 x 8M                    (48 MB) @ +6 MB
//   Qh : [B][H][S][64]             (16 MB) @ +54 MB
//   Kh : [B][H][S][64]             (16 MB) @ +70 MB
//   Vt : [B][H][64][S]             (16 MB) @ +86 MB   (total 102 MB)
// ---------------------------------------------------------------------------
extern "C" void kernel_launch(void* const* d_in, const int* in_sizes, int n_in,
                              void* d_out, int out_size, void* d_ws, size_t ws_size,
                              hipStream_t stream) {
  (void)in_sizes; (void)n_in; (void)out_size; (void)ws_size;
  const float* q  = (const float*)d_in[0];
  const float* k  = (const float*)d_in[1];
  const float* v  = (const float*)d_in[2];
  const float* Wq = (const float*)d_in[3];
  const float* bq = (const float*)d_in[4];
  const float* Wk = (const float*)d_in[5];
  const float* bk = (const float*)d_in[6];
  const float* Wv = (const float*)d_in[7];
  const float* bv = (const float*)d_in[8];
  float* out = (float*)d_out;

  char* ws = (char*)d_ws;
  _Float16* Wt = (_Float16*)(ws);
  _Float16* Xh = (_Float16*)(ws + (size_t)6  * 1024 * 1024);
  _Float16* Qh = (_Float16*)(ws + (size_t)54 * 1024 * 1024);
  _Float16* Kh = (_Float16*)(ws + (size_t)70 * 1024 * 1024);
  _Float16* Vt = (_Float16*)(ws + (size_t)86 * 1024 * 1024);

  wconv_kernel<<<dim3((3u * 1024 * 1024) / 256), dim3(256), 0, stream>>>(Wq, Wk, Wv, Wt);
  xconv_kernel<<<dim3((3u * 8 * 1024 * 1024) / (256 * 4)), dim3(256), 0, stream>>>(q, k, v, Xh);
  proj_kernel<<<dim3(512, 4, 3), dim3(128), 0, stream>>>(Xh, Wt, bq, bk, bv, Qh, Kh, Vt);
  attn_kernel<<<dim3(SS / 128, HH, BB), dim3(256), 0, stream>>>(Qh, Kh, Vt, out);
}
